// GITIII_encoder_last_layer_9646496547380
// MI455X (gfx1250) — matrix operand order repaired and checked by reading, not compiled, not run on hardware
//
#include <hip/hip_runtime.h>

// ---------------------------------------------------------------------------
// GITIII encoder last layer, MI455X (gfx1250, wave32, WMMA bf16 16x16x32).
// Dims: B=64, N=2048, C=512, E=32, E2=64, A=8, IN=512, H=64, MASK_RATIO=10.
//  * Q pre-contracted into Wk -> Wq_eff (B,64,512): 68.7 GFLOP GEMM -> 8.6.
//  * Large GEMMs on v_wmma_f32_16x16x32_bf16 (f32 accumulate).
//  * B-operand weight panels (64x512 bf16 = 64KB) staged into LDS once per
//    block via the Tensor Data Mover (tensor_load_to_lds + s_wait_tensorcnt),
//    with D# row padding (16B per 1024B row) to de-conflict LDS banks.
//  * Softmax over axis=1 via online (max,sum); |tmp| column max via u32
//    atomicMax on non-negative float bit patterns.
// ---------------------------------------------------------------------------

typedef __bf16 bf16_t;
typedef __attribute__((ext_vector_type(16))) __bf16 v16bf;
typedef __attribute__((ext_vector_type(8)))  __bf16 v8bf;
typedef __attribute__((ext_vector_type(8)))  float  v8f;
typedef __attribute__((ext_vector_type(4)))  unsigned v4u;
typedef __attribute__((ext_vector_type(8)))  int      v8i;
typedef __attribute__((ext_vector_type(4)))  int      v4i;

#define WMMA_BF16(A_, B_, C_) \
  __builtin_amdgcn_wmma_f32_16x16x32_bf16(false, (A_), false, (B_), (short)0, (C_), false, false)

static constexpr int Bb = 64, Nn = 2048, Cc = 512, Ee = 32, Aa = 8, INn = 512, Hh = 64;
static constexpr int NM1 = Nn - 1;          // 2047
static constexpr int LDS_LDK = 520;         // 512 + 8 pad elements per row
static constexpr size_t LDS_BYTES = (size_t)64 * LDS_LDK * 2;  // 66560

// ---- workspace layout (bytes) ----
static constexpr size_t OFF_Q     = 0;                                      // f32  B*512
static constexpr size_t OFF_BIAS  = OFF_Q     + (size_t)Bb*512*4;           // f32  B*64
static constexpr size_t OFF_WQEFF = OFF_BIAS  + (size_t)Bb*64*4;            // bf16 B*64*512
static constexpr size_t OFF_VAL   = OFF_WQEFF + (size_t)Bb*64*512*2;        // f32  B*2048*32
static constexpr size_t OFF_HA    = OFF_VAL   + (size_t)Bb*2048*32*4;       // bf16 B*2048*64
static constexpr size_t OFF_HET   = OFF_HA    + (size_t)Bb*2048*64*2;       // bf16 B*2048*64
static constexpr size_t OFF_NTW   = OFF_HET   + (size_t)Bb*2048*64*2;       // bf16 512*512
static constexpr size_t OFF_WA2   = OFF_NTW   + (size_t)512*512*2;          // bf16 512*64
static constexpr size_t OFF_ET2   = OFF_WA2   + (size_t)512*64*2;           // bf16 512*64
static constexpr size_t OFF_CMAX  = OFF_ET2   + (size_t)512*64*2;           // f32  B*512
static constexpr size_t OFF_CSUM  = OFF_CMAX  + (size_t)Bb*512*4;           // f32  B*512
static constexpr size_t OFF_CABS  = OFF_CSUM  + (size_t)Bb*512*4;           // u32  B*512

__device__ __forceinline__ v8f zero8() {
  v8f z = {0.f,0.f,0.f,0.f,0.f,0.f,0.f,0.f};
  return z;
}

// ---------------------------------------------------------------------------
// Stage a 64-row x 512-col bf16 weight panel (row stride 512) from global
// into LDS with row stride 520 (+16B pad per row -> conflict-free B frags).
// Preferred path: Tensor Data Mover (one DMA, TENSORcnt tracked, D# padding).
// Fallback: cooperative ds stores.
// ---------------------------------------------------------------------------
__device__ __forceinline__ void stage_panel_64x512(const bf16_t* gsrc, bf16_t* lds,
                                                   int tid) {
#if __has_builtin(__builtin_amdgcn_tensor_load_to_lds)
  if (tid < 32) {                     // one wave issues the TDM descriptor
    unsigned long long ga = (unsigned long long)(uintptr_t)gsrc;
    v4u g0; v8i g1; v4i g2 = {0,0,0,0}, g3 = {0,0,0,0};
    // D# group 0: count=1 user descriptor; lds_addr = 0 (dynamic-LDS base);
    // 57-bit global address; type=2 ("image").
    g0[0] = 1u;
    g0[1] = 0u;
    g0[2] = (unsigned)ga;
    g0[3] = (unsigned)((ga >> 32) & 0x01FFFFFFu) | (2u << 30);
    // D# group 1: data_size=2B, pad_enable, pad_interval=256 DWORDs (one
    // 512-elem row), pad_amount=4 DWORDs (16B); tensor 512x64, tile 512x64,
    // dim0 stride 512 elements.
    g1[0] = (int)((1u << 16) | (1u << 20) | (7u << 22) | (3u << 25));
    g1[1] = (int)(512u << 16);        // tensor_dim0[15:0] = 512
    g1[2] = (int)(64u  << 16);        // tensor_dim0 hi=0 | tensor_dim1[15:0]=64
    g1[3] = (int)(512u << 16);        // tensor_dim1 hi=0 | tile_dim0 = 512
    g1[4] = 64;                       // tile_dim1 = 64, tile_dim2 = 0
    g1[5] = 512;                      // tensor_dim0_stride = 512
    g1[6] = 0; g1[7] = 0;
#if defined(__clang_major__) && (__clang_major__ >= 23)
    v8i g4 = {0,0,0,0,0,0,0,0};
    __builtin_amdgcn_tensor_load_to_lds(g0, g1, g2, g3, g4, 0);
#else
    __builtin_amdgcn_tensor_load_to_lds(g0, g1, g2, g3, 0);
#endif
    __builtin_amdgcn_s_wait_tensorcnt(0);
  }
  __syncthreads();
#else
  for (int idx = tid; idx < 64*512; idx += 256) {
    int cl = idx >> 9, k = idx & 511;
    lds[cl*LDS_LDK + k] = gsrc[(size_t)cl*512 + k];
  }
  __syncthreads();
#endif
}

// A fragment (16x32 bf16) from an f32 row-major matrix, inline f32->bf16 cvt.
// ISA 7.12.2: lane<16 -> row=lane, K={kk..kk+7, kk+16..kk+23}; lane>=16 ->
// row=lane-16, K shifted by +8.  rmax clamps row (pad-row guard).
__device__ __forceinline__ v16bf load_a_f32(const float* base, int lda, int row0,
                                            int kk, int lane, int rmax) {
  int hf = lane >> 4;
  int r  = row0 + (lane & 15);
  if (r > rmax) r = rmax;
  const float4* p0 = (const float4*)(base + (size_t)r*lda + kk + 8*hf);
  const float4* p1 = (const float4*)(base + (size_t)r*lda + kk + 16 + 8*hf);
  float4 x0 = p0[0], x1 = p0[1], x2 = p1[0], x3 = p1[1];
  v16bf f;
  f[0]=(bf16_t)x0.x;  f[1]=(bf16_t)x0.y;  f[2]=(bf16_t)x0.z;  f[3]=(bf16_t)x0.w;
  f[4]=(bf16_t)x1.x;  f[5]=(bf16_t)x1.y;  f[6]=(bf16_t)x1.z;  f[7]=(bf16_t)x1.w;
  f[8]=(bf16_t)x2.x;  f[9]=(bf16_t)x2.y;  f[10]=(bf16_t)x2.z; f[11]=(bf16_t)x2.w;
  f[12]=(bf16_t)x3.x; f[13]=(bf16_t)x3.y; f[14]=(bf16_t)x3.z; f[15]=(bf16_t)x3.w;
  return f;
}

// A fragment from a bf16 row-major matrix (rows pre-padded, no clamp needed).
__device__ __forceinline__ v16bf load_a_bf16(const bf16_t* base, int lda, int row0,
                                             int kk, int lane) {
  int hf = lane >> 4;
  int r  = row0 + (lane & 15);
  const v8bf* p0 = (const v8bf*)(base + (size_t)r*lda + kk + 8*hf);
  const v8bf* p1 = (const v8bf*)(base + (size_t)r*lda + kk + 16 + 8*hf);
  v8bf x0 = *p0, x1 = *p1;
  v16bf f;
#pragma unroll
  for (int i = 0; i < 8; ++i) { f[i] = x0[i]; f[8+i] = x1[i]; }
  return f;
}

// B fragment (32x16) from bf16 W[c][k] row-major with leading dim ldk:
// lane<16 -> col=c0+lane, K={kk..kk+7, kk+16..kk+23}; lane>=16 -> col, K shifted by +8.
__device__ __forceinline__ v16bf load_b_bf16(const bf16_t* w, int ldk, int c0,
                                             int kk, int lane) {
  int c = c0 + (lane & 15);
  int k = kk + 8*(lane >> 4);
  const v8bf* p = (const v8bf*)(w + (size_t)c*ldk + k);
  v8bf x0 = p[0], x1 = p[2];
  v16bf f;
#pragma unroll
  for (int i = 0; i < 8; ++i) { f[i] = x0[i]; f[8+i] = x1[i]; }
  return f;
}

__device__ __forceinline__ float gelu_exact(float x) {
  return 0.5f * x * (1.f + erff(x * 0.70710678118654752f));
}
__device__ __forceinline__ float selu_f(float x) {
  const float sc = 1.0507009873554805f, al = 1.6732632423543772f;
  return (x > 0.f) ? sc * x : sc * al * (__expf(x) - 1.f);
}

// ---------------------------------------------------------------------------
// K-prep: f32 weights -> bf16 copies in ws (nt_w 512x512, Wa_w2/et_w2 512x64)
// ---------------------------------------------------------------------------
__global__ void prep_w_kernel(const float* nt_w, const float* wa_w2, const float* et_w2,
                              bf16_t* nt_bf, bf16_t* wa_bf, bf16_t* et_bf) {
  int gid = blockIdx.x * blockDim.x + threadIdx.x;
  int stride = gridDim.x * blockDim.x;
  for (int i = gid; i < 512*512; i += stride) nt_bf[i] = (bf16_t)nt_w[i];
  for (int i = gid; i < 512*64;  i += stride) wa_bf[i] = (bf16_t)wa_w2[i];
  for (int i = gid; i < 512*64;  i += stride) et_bf[i] = (bf16_t)et_w2[i];
}

// ---------------------------------------------------------------------------
// Q[b, a*64+e] = node[b,0,:] . Wq_w[a*64+e,:] + Wq_b   (B x 512, tiny)
// ---------------------------------------------------------------------------
__global__ void q_kernel(const float* node, const float* Wq_w, const float* Wq_b, float* Q) {
  int b = blockIdx.x;
  const float* x = node + (size_t)b * Nn * Cc;   // row n=0
  for (int o = threadIdx.x; o < 512; o += blockDim.x) {
    const float* w = Wq_w + (size_t)o * Cc;
    float s = 0.f;
#pragma unroll 8
    for (int k = 0; k < Cc; ++k) s += x[k] * w[k];
    Q[(size_t)b*512 + o] = s + Wq_b[o];
  }
}

// ---------------------------------------------------------------------------
// Wq_eff[b,e,k] = sum_a Q[b,a,e] * Wk_w[a*64+e, k]  (stored bf16)
// bias_eff[b,e] = sum_a Q[b,a,e] * Wk_b[a*64+e]
// ---------------------------------------------------------------------------
__global__ void wqeff_kernel(const float* Q, const float* Wk_w, const float* Wk_b,
                             bf16_t* wq_eff, float* bias_eff) {
  int b = blockIdx.x;
  const float* q = Q + (size_t)b * 512;
  for (int idx = threadIdx.x; idx < 64*512; idx += blockDim.x) {
    int e = idx >> 9, k = idx & 511;
    float s = 0.f;
#pragma unroll
    for (int a = 0; a < Aa; ++a)
      s += q[a*64 + e] * Wk_w[(size_t)(a*64 + e)*Cc + k];
    wq_eff[(size_t)b*64*512 + idx] = (bf16_t)s;
  }
  if (threadIdx.x < 64) {
    int e = threadIdx.x;
    float s = 0.f;
#pragma unroll
    for (int a = 0; a < Aa; ++a) s += q[a*64 + e] * Wk_b[a*64 + e];
    bias_eff[(size_t)b*64 + e] = s;
  }
}

// ---------------------------------------------------------------------------
// val_kernel: QK = node @ Wq_eff.T (+bias) via WMMA (B panel TDM-staged in
// LDS), fused ew/rho/selu.  grid (16 ntiles, B), 8 waves x (16 rows x 64 c).
// ---------------------------------------------------------------------------
__global__ void val_kernel(const float* node, const float* edge,
                           const float* emb1, const float* emb2,
                           const float* Wew, const bf16_t* wq_eff,
                           const float* bias_eff, float* valout) {
  extern __shared__ char smem_raw[];
  bf16_t* wtile = (bf16_t*)smem_raw;             // [64][520] bf16

  int b = blockIdx.y;
  int lane = threadIdx.x & 31, wave = threadIdx.x >> 5;
  int row0 = blockIdx.x * 128 + wave * 16;       // n-row base
  const float* nbase = node + (size_t)b * Nn * Cc;

  stage_panel_64x512(wq_eff + (size_t)b * 64 * 512, wtile, threadIdx.x);

  v8f acc[4] = {zero8(), zero8(), zero8(), zero8()};
  for (int kk = 0; kk < Cc; kk += 32) {
    if (kk + 64 < Cc)
      __builtin_prefetch(nbase + (size_t)(row0 + (lane & 15))*Cc + kk + 64, 0, 1);
    v16bf a = load_a_f32(nbase, Cc, row0, kk, lane, Nn - 1);
#pragma unroll
    for (int cg = 0; cg < 4; ++cg) {
      v16bf bm = load_b_bf16(wtile, LDS_LDK, cg*16, kk, lane);   // ds reads
      acc[cg] = WMMA_BF16(a, bm, acc[cg]);
    }
  }

  int hf = lane >> 4, ln = lane & 15;
#pragma unroll
  for (int cg = 0; cg < 2; ++cg) {
    int e = cg*16 + ln;
    float blo = bias_eff[(size_t)b*64 + e];
    float bhi = bias_eff[(size_t)b*64 + e + 32];
    const float* wew = Wew + (size_t)e * Ee;
#pragma unroll
    for (int i = 0; i < 8; ++i) {
      int r = row0 + 8*hf + i;
      const float* erow = edge + ((size_t)b*Nn + r) * Ee;
      float ew = 0.f;
#pragma unroll
      for (int k2 = 0; k2 < Ee; ++k2) ew += erow[k2] * wew[k2];
      float qlo = acc[cg][i]   + blo;
      float qhi = acc[cg+2][i] + bhi;
      float x = qlo * ew;
      float rh = sqrtf(fmaxf(x, 0.f)) - sqrtf(fmaxf(-x, 0.f));
      size_t ei = ((size_t)b*Nn + r) * Ee + e;
      float v = rh + qhi * emb1[ei] + emb2[ei];
      valout[ei] = selu_f(v);
    }
  }
}

// ---------------------------------------------------------------------------
// h_kernel: h_a = gelu(e@Wa_w1.T), h_et = gelu(e@et_w1.T + et_b1), bf16,
// padded to 2048 rows (row j holds e = val[j+1]).  grid (512, B), 4 rows/blk.
// ---------------------------------------------------------------------------
__global__ void h_kernel(const float* val, const float* Wa_w1,
                         const float* et_w1, const float* et_b1,
                         bf16_t* h_a, bf16_t* h_et) {
  int b = blockIdx.y;
  int j = blockIdx.x * 4 + (threadIdx.x >> 6);
  int h = threadIdx.x & 63;
  size_t o = ((size_t)b*Nn + j) * Hh + h;
  if (j >= NM1) { h_a[o] = (bf16_t)0.f; h_et[o] = (bf16_t)0.f; return; }
  const float* e = val + ((size_t)b*Nn + (j + 1)) * Ee;
  const float* w1 = Wa_w1 + (size_t)h * Ee;
  const float* w2 = et_w1 + (size_t)h * Ee;
  float s1 = 0.f, s2 = 0.f;
#pragma unroll
  for (int k = 0; k < Ee; ++k) { float ev = e[k]; s1 += ev*w1[k]; s2 += ev*w2[k]; }
  h_a[o]  = (bf16_t)gelu_exact(s1);
  h_et[o] = (bf16_t)gelu_exact(s2 + et_b1[h]);
}

// ---------------------------------------------------------------------------
// stats_kernel: logits = h_a @ Wa_w2.T, online softmax stats over j (axis=1).
// grid (32 ctiles of 16, B).  Also zero-inits the |tmp| column-max.
// ---------------------------------------------------------------------------
__global__ void stats_kernel(const bf16_t* h_a, const bf16_t* wa_bf,
                             float* colmax, float* colsum, unsigned* colabs) {
  int b = blockIdx.y;
  int c0 = blockIdx.x * 16;
  int lane = threadIdx.x & 31, wave = threadIdx.x >> 5;
  const bf16_t* ha = h_a + (size_t)b * Nn * Hh;

  v16bf bm0 = load_b_bf16(wa_bf, Hh, c0, 0,  lane);
  v16bf bm1 = load_b_bf16(wa_bf, Hh, c0, 32, lane);

  float m = -3.4e38f, s = 0.f;
  int hf = lane >> 4;
  for (int jt = 0; jt < 16; ++jt) {
    int row0 = jt * 128 + wave * 16;
    v16bf a0 = load_a_bf16(ha, Hh, row0, 0,  lane);
    v16bf a1 = load_a_bf16(ha, Hh, row0, 32, lane);
    v8f c = zero8();
    c = WMMA_BF16(a0, bm0, c);
    c = WMMA_BF16(a1, bm1, c);
#pragma unroll
    for (int i = 0; i < 8; ++i) {
      int j = row0 + 8*hf + i;
      if (j < NM1) {
        float x = c[i];
        if (x > m) { s = s * __expf(m - x) + 1.f; m = x; }
        else       { s += __expf(x - m); }
      }
    }
  }
  float m2 = __shfl_xor(m, 16, 32);
  float s2 = __shfl_xor(s, 16, 32);
  float M = fmaxf(m, m2);
  s = s * __expf(m - M) + s2 * __expf(m2 - M);
  m = M;

  __shared__ float sm[8][16];
  __shared__ float ss[8][16];
  if (lane < 16) { sm[wave][lane] = m; ss[wave][lane] = s; }
  __syncthreads();
  if (threadIdx.x < 16) {
    int c = threadIdx.x;
    float gm = sm[0][c], gs = ss[0][c];
#pragma unroll
    for (int w = 1; w < 8; ++w) {
      float wm = sm[w][c], wsv = ss[w][c];
      float nm = fmaxf(gm, wm);
      gs = gs * __expf(gm - nm) + wsv * __expf(wm - nm);
      gm = nm;
    }
    colmax[(size_t)b*INn + c0 + c] = gm;
    colsum[(size_t)b*INn + c0 + c] = gs;
    colabs[(size_t)b*INn + c0 + c] = 0u;   // init for fused_kernel atomics
  }
}

// ---------------------------------------------------------------------------
// fused_kernel: node_t (K=512 WMMA, B panel TDM-staged) + edge_t + logits,
// alphas, tmp = (node_t+edge_t)*alpha -> d_out tmp region, atomic |tmp| max.
// grid (8 ctiles of 64, 16 jtiles, B); wave -> 16 rows x 64 cols.
// ---------------------------------------------------------------------------
__global__ void fused_kernel(const float* node, const bf16_t* h_a, const bf16_t* h_et,
                             const bf16_t* nt_bf, const bf16_t* wa_bf, const bf16_t* et_bf,
                             const float* nt_b, const float* et_b2,
                             const float* colmax, const float* colsum,
                             unsigned* colabs, float* tmp_out) {
  extern __shared__ char smem_raw[];
  bf16_t* wtile = (bf16_t*)smem_raw;             // [64][520] bf16

  int b = blockIdx.z;
  int c0 = blockIdx.x * 64;
  int lane = threadIdx.x & 31, wave = threadIdx.x >> 5;
  int row0 = blockIdx.y * 128 + wave * 16;               // j-row base
  const float*  nbase = node + ((size_t)b * Nn + 1) * Cc; // row n = 1 + j
  const bf16_t* ha = h_a  + (size_t)b * Nn * Hh;
  const bf16_t* he = h_et + (size_t)b * Nn * Hh;

  stage_panel_64x512(nt_bf + (size_t)c0 * 512, wtile, threadIdx.x);

  v8f accN[4] = {zero8(), zero8(), zero8(), zero8()};
  v8f accL[4] = {zero8(), zero8(), zero8(), zero8()};
  v8f accE[4] = {zero8(), zero8(), zero8(), zero8()};

  for (int kk = 0; kk < Cc; kk += 32) {
    if (kk + 64 < Cc)
      __builtin_prefetch(nbase + (size_t)(row0 + (lane & 15))*Cc + kk + 64, 0, 1);
    v16bf a = load_a_f32(nbase, Cc, row0, kk, lane, NM1 - 1);  // clamp pad row
#pragma unroll
    for (int cg = 0; cg < 4; ++cg) {
      v16bf bm = load_b_bf16(wtile, LDS_LDK, cg*16, kk, lane);  // ds reads
      accN[cg] = WMMA_BF16(a, bm, accN[cg]);
    }
  }
#pragma unroll
  for (int kk = 0; kk < Hh; kk += 32) {
    v16bf va = load_a_bf16(ha, Hh, row0, kk, lane);
    v16bf ve = load_a_bf16(he, Hh, row0, kk, lane);
#pragma unroll
    for (int cg = 0; cg < 4; ++cg) {
      v16bf bl = load_b_bf16(wa_bf, Hh, c0 + cg*16, kk, lane);
      v16bf bt = load_b_bf16(et_bf, Hh, c0 + cg*16, kk, lane);
      accL[cg] = WMMA_BF16(va, bl, accL[cg]);
      accE[cg] = WMMA_BF16(ve, bt, accE[cg]);
    }
  }

  int hf = lane >> 4, ln = lane & 15;
#pragma unroll
  for (int cg = 0; cg < 4; ++cg) {
    int c = c0 + cg*16 + ln;
    float cm  = colmax[(size_t)b*INn + c];
    float inv = 1.f / colsum[(size_t)b*INn + c];
    float bN  = nt_b[c], bE = et_b2[c];
    float lmax = 0.f;
#pragma unroll
    for (int i = 0; i < 8; ++i) {
      int j = row0 + 8*hf + i;
      if (j < NM1) {
        float alpha = __expf(accL[cg][i] - cm) * inv;
        float t = (accN[cg][i] + bN + accE[cg][i] + bE) * alpha;
        tmp_out[((size_t)b*NM1 + j) * INn + c] = t;
        lmax = fmaxf(lmax, fabsf(t));
      }
    }
    atomicMax(&colabs[(size_t)b*INn + c], __float_as_uint(lmax));
  }
}

// ---------------------------------------------------------------------------
// mask_sum: thr = colmax/10; mask tmp in place; out[b,c] = sum_j tmp.
// ---------------------------------------------------------------------------
__global__ void mask_sum_kernel(const unsigned* colabs, float* tmp, float* out) {
  int b = blockIdx.y;
  int c = blockIdx.x * 256 + threadIdx.x;
  float thr = __uint_as_float(colabs[(size_t)b*INn + c]) * 0.1f;
  float s = 0.f;
  size_t base = (size_t)b * NM1 * INn + c;
  for (int j = 0; j < NM1; ++j) {
    float v = tmp[base + (size_t)j * INn];
    float mv = (fabsf(v) >= thr) ? v : 0.f;
    tmp[base + (size_t)j * INn] = mv;
    s += mv;
  }
  out[(size_t)b*INn + c] = s;
}

// ---------------------------------------------------------------------------
extern "C" void kernel_launch(void* const* d_in, const int* in_sizes, int n_in,
                              void* d_out, int out_size, void* d_ws, size_t ws_size,
                              hipStream_t stream) {
  const float* node  = (const float*)d_in[0];
  const float* edge  = (const float*)d_in[1];
  const float* emb1  = (const float*)d_in[2];
  const float* emb2  = (const float*)d_in[3];
  const float* Wq_w  = (const float*)d_in[4];
  const float* Wq_b  = (const float*)d_in[5];
  const float* Wk_w  = (const float*)d_in[6];
  const float* Wk_b  = (const float*)d_in[7];
  const float* Wew_w = (const float*)d_in[8];
  const float* Wa_w1 = (const float*)d_in[9];
  const float* Wa_w2 = (const float*)d_in[10];
  const float* nt_w  = (const float*)d_in[11];
  const float* nt_b  = (const float*)d_in[12];
  const float* et_w1 = (const float*)d_in[13];
  const float* et_b1 = (const float*)d_in[14];
  const float* et_w2 = (const float*)d_in[15];
  const float* et_b2 = (const float*)d_in[16];

  char* ws = (char*)d_ws;
  float*    Q        = (float*)   (ws + OFF_Q);
  float*    bias_eff = (float*)   (ws + OFF_BIAS);
  bf16_t*   wq_eff   = (bf16_t*)  (ws + OFF_WQEFF);
  float*    val      = (float*)   (ws + OFF_VAL);
  bf16_t*   h_a      = (bf16_t*)  (ws + OFF_HA);
  bf16_t*   h_et     = (bf16_t*)  (ws + OFF_HET);
  bf16_t*   nt_bf    = (bf16_t*)  (ws + OFF_NTW);
  bf16_t*   wa_bf    = (bf16_t*)  (ws + OFF_WA2);
  bf16_t*   et_bf    = (bf16_t*)  (ws + OFF_ET2);
  float*    cmax     = (float*)   (ws + OFF_CMAX);
  float*    csum     = (float*)   (ws + OFF_CSUM);
  unsigned* cabs     = (unsigned*)(ws + OFF_CABS);

  float* out = (float*)d_out;                 // (B, 512)
  float* tmp = out + (size_t)Bb * INn;        // (B, 2047, 512)

  prep_w_kernel <<<1024, 256, 0, stream>>>(nt_w, Wa_w2, et_w2, nt_bf, wa_bf, et_bf);
  q_kernel      <<<Bb,   256, 0, stream>>>(node, Wq_w, Wq_b, Q);
  wqeff_kernel  <<<Bb,   256, 0, stream>>>(Q, Wk_w, Wk_b, wq_eff, bias_eff);
  val_kernel    <<<dim3(16, Bb), 256, LDS_BYTES, stream>>>(node, edge, emb1, emb2,
                                                           Wew_w, wq_eff, bias_eff, val);
  h_kernel      <<<dim3(512, Bb),   256, 0, stream>>>(val, Wa_w1, et_w1, et_b1, h_a, h_et);
  stats_kernel  <<<dim3(32, Bb),    256, 0, stream>>>(h_a, wa_bf, cmax, csum, cabs);
  fused_kernel  <<<dim3(8, 16, Bb), 256, LDS_BYTES, stream>>>(node, h_a, h_et, nt_bf,
                                                              wa_bf, et_bf, nt_b, et_b2,
                                                              cmax, csum, cabs, tmp);
  mask_sum_kernel<<<dim3(2, Bb),    256, 0, stream>>>(cabs, tmp, out);
}
